// MTREncoder_87428354278002
// MI455X (gfx1250) — matrix operand order repaired and checked by reading, not compiled, not run on hardware
//
#include <hip/hip_runtime.h>
#include <hip/hip_bf16.h>

// ---------------------------------------------------------------------------
// MTR encoder for MI455X (gfx1250, wave32).
// All GEMM-shaped work goes through v_wmma_f32_16x16x32_f16 (f16 in, f32 acc).
// GEMM: 64x64 macro-tile per 4-wave workgroup, weight tile double-buffered in
// LDS via global_load_async_to_lds_b128, 4x A-operand reuse per wave.
// Softmax / layernorm / pooling / masks in f32 VALU.
// ---------------------------------------------------------------------------

typedef __attribute__((ext_vector_type(16))) _Float16 half16;
typedef __attribute__((ext_vector_type(8)))  float    float8;
typedef int aint4 __attribute__((vector_size(16)));   // GCC-vector int4 for async builtin

#define AS1 __attribute__((address_space(1)))
#define AS3 __attribute__((address_space(3)))

#if defined(__HIP_DEVICE_COMPILE__)
#if __has_builtin(__builtin_amdgcn_global_load_async_to_lds_b128) && \
    __has_builtin(__builtin_amdgcn_s_wait_asynccnt)
#define ASYNC_LDS 1
#else
#define ASYNC_LDS 0
#endif
#else
#define ASYNC_LDS 0
#endif

#define NB   8
#define NOBJ 48
#define TT   11
#define AGA  29
#define NTY  16
#define PERT 48
#define NPTS 20
#define MPA  9
#define NPOLY (NTY*PERT)          // 768
#define DD   256
#define HH   8
#define DH   32
#define LL   2
#define FFN  1024
#define NTOK 1296                  // 528 obj + 768 map
#define AGROWS (NB*NOBJ*TT)        // 4224
#define MPROWS (NB*PERT*NPTS)      // 7680 per type
#define TROWS  (NB*NTOK)           // 10368
#define OUTTOK 8976                // 528 + 768*11

// ---------------------------------------------------------------------------
// WMMA GEMM, 64x64 macro-tile per 128-thread block (4 waves).
// out = act(X[M,K] @ Wt[N,K]^T + bias) * rmask.  K mult of 32; M,N mult of 64.
// Weight tile (64 cols x 32 K, 4KB) staged to LDS, double buffered; each wave
// computes one 16-row strip and reuses its A fragment across 4 column tiles.
// ---------------------------------------------------------------------------
template<bool RELU>
__global__ void __launch_bounds__(128)
gemm64_kernel(const _Float16* __restrict__ X, long long xStride,
              const _Float16* __restrict__ W, long long wStride,
              const float* __restrict__ bias, long long bStride,
              const float* __restrict__ rmask, long long mStride,
              float* __restrict__ outF, _Float16* __restrict__ outH,
              long long oStride, int M, int N, int K)
{
    __shared__ __align__(16) _Float16 Bs[2][64][32];
    int tid = threadIdx.x;
    int lane = tid & 31, wave = tid >> 5;
    int lo = lane >> 4, nn = lane & 15;
    int row0 = blockIdx.x * 64 + wave * 16;
    int col0 = blockIdx.y * 64;
    int bb = blockIdx.z;
    const _Float16* ap = X + (long long)bb * xStride + (long long)(row0 + nn) * K;
    // staging assignment: thread -> (weight column sc, 16-half segment sp)
    int sc = tid >> 1;
    int sp = (tid & 1) * 16;
    const _Float16* sg = W + (long long)bb * wStride + (long long)(col0 + sc) * K + sp;

    float8 acc[4] = {{}, {}, {}, {}};

    auto stage = [&](int buf, int k0) {
#if ASYNC_LDS
        __builtin_amdgcn_global_load_async_to_lds_b128(
            (AS1 aint4*)(sg + k0), (AS3 aint4*)&Bs[buf][sc][sp], 0, 0);
        __builtin_amdgcn_global_load_async_to_lds_b128(
            (AS1 aint4*)(sg + k0 + 8), (AS3 aint4*)&Bs[buf][sc][sp + 8], 0, 0);
#else
        const uint4* s = (const uint4*)(sg + k0);
        uint4* d = (uint4*)&Bs[buf][sc][sp];
        d[0] = s[0];
        d[1] = s[1];
#endif
    };

    stage(0, 0);
    int nk = K >> 5;
    for (int kt = 0; kt < nk; ++kt) {
#if ASYNC_LDS
        __builtin_amdgcn_s_wait_asynccnt(0);
#endif
        __syncthreads();
        if (kt + 1 < nk) stage((kt + 1) & 1, (kt + 1) << 5);
        int k0 = kt << 5;
        if (k0 + 32 < K) __builtin_prefetch(ap + k0 + 32, 0, 1);  // global_prefetch
        half16 a;
#pragma unroll
        for (int j = 0; j < 8; ++j) {
            a[j]     = ap[k0 + lo*16 + j];         // K = lo*16+j
            a[j + 8] = ap[k0 + lo*16 + 8 + j];     // K = lo*16+8+j
        }
        const _Float16* bsrc = &Bs[kt & 1][0][0];
#pragma unroll
        for (int t = 0; t < 4; ++t) {
            half16 b;
#pragma unroll
            for (int j = 0; j < 16; ++j)
                b[j] = bsrc[(t * 16 + nn) * 32 + lo * 16 + j];  // K = lo*16+j
            acc[t] = __builtin_amdgcn_wmma_f32_16x16x32_f16(false, a, false, b,
                                                            (short)0, acc[t],
                                                            false, false);
        }
        __syncthreads();   // protect current buffer from next iteration's stage
    }

#pragma unroll
    for (int t = 0; t < 4; ++t) {
        int cb = col0 + t * 16;
#pragma unroll
        for (int r = 0; r < 8; ++r) {
            int m = row0 + lo * 8 + r;
            float v = acc[t][r];
            if (bias)  v += bias[bb * bStride + cb + nn];
            if (RELU)  v = fmaxf(v, 0.f);
            if (rmask) v *= rmask[bb * mStride + m];
            long long o = (long long)bb * oStride + (long long)m * N + cb + nn;
            if (outF) outF[o] = v;
            if (outH) outH[o] = (_Float16)v;
        }
    }
}

// ---------------------------------------------------------------------------
// Flash attention: one wave per (q-tile of 16, head, batch). DH=32 -> one WMMA
// per score tile; 81 key chunks of 16 (upper 16 K slots of P zero-padded).
// ---------------------------------------------------------------------------
__global__ void __launch_bounds__(32)
attn_kernel(const _Float16* __restrict__ Q, const _Float16* __restrict__ Kt,
            const _Float16* __restrict__ V, const float* __restrict__ tmask,
            _Float16* __restrict__ O)
{
    int lane = threadIdx.x & 31;
    int lo = lane >> 4, nn = lane & 15;
    int q0 = blockIdx.x * 16, h = blockIdx.y, b = blockIdx.z;
    long long base = (long long)b * NTOK * DD + h * DH;
    const _Float16* Qb = Q + base;
    const _Float16* Kb = Kt + base;
    const _Float16* Vb = V + base;
    const float* mb = tmask + (long long)b * NTOK;

    half16 aq;
    {
        const _Float16* p = Qb + (long long)(q0 + nn) * DD;
#pragma unroll
        for (int j = 0; j < 8; ++j) { aq[j] = p[lo*16 + j]; aq[j + 8] = p[lo*16 + 8 + j]; }
    }
    float8 acc0 = {}, acc1 = {};
    float Mr[8], Sr[8];
#pragma unroll
    for (int r = 0; r < 8; ++r) { Mr[r] = -1e30f; Sr[r] = 0.f; }

    __shared__ _Float16 Pt[16 * 32];
    for (int i = lane; i < 16 * 16; i += 32) Pt[(i >> 4) * 32 + 16 + (i & 15)] = (_Float16)0.f;
    __syncthreads();

    const float scale = 0.17677669529663687f;   // 1/sqrt(32)
    for (int k0 = 0; k0 < NTOK; k0 += 16) {
        half16 bk;
        {
            const _Float16* p = Kb + (long long)(k0 + nn) * DD + lo * 16;
#pragma unroll
            for (int j = 0; j < 16; ++j) bk[j] = p[j];   // B[dh=lo*16+j][key=nn]
        }
        float8 sc = {};
        sc = __builtin_amdgcn_wmma_f32_16x16x32_f16(false, aq, false, bk,
                                                    (short)0, sc, false, false);
        float madd = (mb[k0 + nn] > 0.5f) ? 0.f : -1e9f;
        float pv[8];
#pragma unroll
        for (int r = 0; r < 8; ++r) {
            float s = sc[r] * scale + madd;
            float t = s;
            t = fmaxf(t, __shfl_xor(t, 1, 16));
            t = fmaxf(t, __shfl_xor(t, 2, 16));
            t = fmaxf(t, __shfl_xor(t, 4, 16));
            t = fmaxf(t, __shfl_xor(t, 8, 16));
            float nM = fmaxf(Mr[r], t);
            float f = __expf(Mr[r] - nM);
            float e = __expf(s - nM);
            float rs = e;
            rs += __shfl_xor(rs, 1, 16);
            rs += __shfl_xor(rs, 2, 16);
            rs += __shfl_xor(rs, 4, 16);
            rs += __shfl_xor(rs, 8, 16);
            Sr[r] = Sr[r] * f + rs;
            Mr[r] = nM;
            acc0[r] *= f; acc1[r] *= f;
            pv[r] = e;
        }
        __syncthreads();
#pragma unroll
        for (int r = 0; r < 8; ++r) Pt[(lo*8 + r) * 32 + nn] = (_Float16)pv[r];
        __syncthreads();
        half16 aP;
        {
            const _Float16* p = Pt + nn * 32;
#pragma unroll
            for (int j = 0; j < 8; ++j) { aP[j] = p[lo*16 + j]; aP[j + 8] = p[lo*16 + 8 + j]; }
        }
        half16 bv0, bv1;
#pragma unroll
        for (int j = 0; j < 16; ++j) {
            // B k-index = lo*16+j -> only lo==0 rows are real (cols 16..31 of P are 0)
            bv0[j] = (lo == 0) ? Vb[(long long)(k0 + j) * DD + nn]      : (_Float16)0.f;
            bv1[j] = (lo == 0) ? Vb[(long long)(k0 + j) * DD + 16 + nn] : (_Float16)0.f;
        }
        acc0 = __builtin_amdgcn_wmma_f32_16x16x32_f16(false, aP, false, bv0,
                                                      (short)0, acc0, false, false);
        acc1 = __builtin_amdgcn_wmma_f32_16x16x32_f16(false, aP, false, bv1,
                                                      (short)0, acc1, false, false);
    }
#pragma unroll
    for (int r = 0; r < 8; ++r) {
        int q = q0 + lo*8 + r;
        float inv = 1.f / fmaxf(Sr[r], 1e-30f);
        long long o = (long long)b * NTOK * DD + (long long)q * DD + h * DH;
        O[o + nn]      = (_Float16)(acc0[r] * inv);
        O[o + 16 + nn] = (_Float16)(acc1[r] * inv);
    }
}

// ---------------------------------------------------------------------------
// Elementwise / glue kernels
// ---------------------------------------------------------------------------
__global__ void convert_wT_kernel(const float* __restrict__ W, _Float16* __restrict__ Wt,
                                  int K, int N, int Kp, long long total)
{
    long long i = (long long)blockIdx.x * blockDim.x + threadIdx.x;
    if (i >= total) return;
    int kp = (int)(i % Kp);
    long long t = i / Kp;
    int n = (int)(t % N);
    long long bb = t / N;
    Wt[i] = (_Float16)((kp < K) ? W[(bb * K + kp) * N + n] : 0.f);
}

__global__ void pack_agent_kernel(const float* __restrict__ trajs,
                                  const unsigned char* __restrict__ mask,
                                  _Float16* __restrict__ X, float* __restrict__ am,
                                  long long total)
{
    long long i = (long long)blockIdx.x * blockDim.x + threadIdx.x;
    if (i >= total) return;                       // total = AGROWS*32
    int j = (int)(i & 31);
    long long r = i >> 5;
    float mv = mask[r] ? 1.f : 0.f;
    float v = (j < AGA) ? trajs[r * AGA + j] : ((j == AGA) ? mv : 0.f);
    X[i] = (_Float16)v;
    if (j == 0) am[r] = mv;
}

__global__ void pack_map_kernel(const float* __restrict__ polys,
                                const unsigned char* __restrict__ mask,
                                _Float16* __restrict__ X, float* __restrict__ mm,
                                long long total)
{
    long long i = (long long)blockIdx.x * blockDim.x + threadIdx.x;
    if (i >= total) return;                       // total = NTY*MPROWS*32
    int j = (int)(i & 31);
    long long r = i >> 5;                         // r over NTY*MPROWS (type-major)
    int ty = (int)(r / MPROWS);
    int rr = (int)(r % MPROWS);
    int pt = rr % NPTS;
    int rowb = rr / NPTS;                         // b*48+pp
    int pp = rowb % PERT, b = rowb / PERT;
    long long src = ((long long)b * NPOLY + ty * PERT + pp) * NPTS + pt;
    X[i] = (_Float16)((j < MPA) ? polys[src * MPA + j] : 0.f);
    if (j == 0) mm[r] = mask[src] ? 1.f : 0.f;
}

__global__ void mvalid_kernel(const float* __restrict__ mm, float* __restrict__ mv,
                              long long total)
{
    long long i = (long long)blockIdx.x * blockDim.x + threadIdx.x;
    if (i >= total) return;                       // total = NTY*NB*PERT
    int ty = (int)(i / (NB * PERT));
    int rr = (int)(i % (NB * PERT));
    float v = 0.f;
    for (int pt = 0; pt < NPTS; ++pt)
        v = fmaxf(v, mm[(long long)ty * MPROWS + rr * NPTS + pt]);
    mv[i] = v;
}

__global__ void poolmax_kernel(const _Float16* __restrict__ in, _Float16* __restrict__ out,
                               int G, int N, long long totalOut)
{
    long long i = (long long)blockIdx.x * blockDim.x + threadIdx.x;
    if (i >= totalOut) return;
    int n = (int)(i % N);
    long long ro = i / N;
    float m = -1e30f;
    for (int g = 0; g < G; ++g)
        m = fmaxf(m, (float)in[(ro * G + g) * (long long)N + n]);
    out[i] = (_Float16)m;
}

__global__ void concat_kernel(const _Float16* __restrict__ A, const _Float16* __restrict__ P,
                              _Float16* __restrict__ out, int G, int N1, long long total)
{
    long long i = (long long)blockIdx.x * blockDim.x + threadIdx.x;
    if (i >= total) return;
    int N2 = 2 * N1;
    int j = (int)(i % N2);
    long long r = i / N2;
    out[i] = (j < N1) ? A[r * N1 + j] : P[(r / G) * (long long)N1 + (j - N1)];
}

__global__ void scatter_obj_kernel(const float* __restrict__ of, float* __restrict__ tok,
                                   long long total)
{
    long long i = (long long)blockIdx.x * blockDim.x + threadIdx.x;
    if (i >= total) return;                       // total = AGROWS*DD
    int d = (int)(i & 255);
    long long r = i >> 8;
    int b = (int)(r / (NOBJ * TT)), q = (int)(r % (NOBJ * TT));
    tok[((long long)b * NTOK + q) * DD + d] = of[i];
}

__global__ void scatter_map_kernel(const float* __restrict__ mf, float* __restrict__ tok,
                                   long long total)
{
    long long i = (long long)blockIdx.x * blockDim.x + threadIdx.x;
    if (i >= total) return;                       // total = NTY*NB*PERT*DD
    int d = (int)(i & 255);
    long long s = i >> 8;
    int ty = (int)(s / (NB * PERT));
    int rr = (int)(s % (NB * PERT));
    int b = rr / PERT, pp = rr % PERT;
    tok[((long long)b * NTOK + NOBJ * TT + ty * PERT + pp) * DD + d] = mf[i];
}

__global__ void tokmask_kernel(const float* __restrict__ am, const float* __restrict__ mv,
                               float* __restrict__ tm, long long total)
{
    long long i = (long long)blockIdx.x * blockDim.x + threadIdx.x;
    if (i >= total) return;                       // total = NB*NTOK
    int tok = (int)(i % NTOK), b = (int)(i / NTOK);
    if (tok < NOBJ * TT) tm[i] = am[(long long)b * NOBJ * TT + tok];
    else {
        int q = tok - NOBJ * TT;
        tm[i] = mv[(q / PERT) * (NB * PERT) + b * PERT + (q % PERT)];
    }
}

__global__ void posembed_kernel(const float* __restrict__ op, const float* __restrict__ mp,
                                float* __restrict__ pe, long long total)
{
    long long i = (long long)blockIdx.x * blockDim.x + threadIdx.x;
    if (i >= total) return;                       // total = NB*NTOK*DD
    int d = (int)(i & 255);
    long long t = i >> 8;
    int tok = (int)(t % NTOK), b = (int)(t / NTOK);
    const float* p = (tok < NOBJ * TT)
        ? op + ((long long)b * NOBJ * TT + tok) * 3
        : mp + ((long long)b * NPOLY + (tok - NOBJ * TT)) * 3;
    float v = (d < 128) ? p[1] : p[0];            // y-embed first, then x
    int j = d & 127, k = j >> 1;
    float dt = powf(10000.f, (float)k / 64.f);
    float th = v * 6.283185307179586f / dt;
    pe[i] = (j & 1) ? cosf(th) : sinf(th);
}

__global__ void prep_kernel(const float* __restrict__ x, const float* __restrict__ pos,
                            _Float16* __restrict__ x16, _Float16* __restrict__ q16,
                            long long total)
{
    long long i = (long long)blockIdx.x * blockDim.x + threadIdx.x;
    if (i >= total) return;
    float v = x[i];
    x16[i] = (_Float16)v;
    q16[i] = (_Float16)(v + pos[i]);
}

__global__ void __launch_bounds__(256)
add_ln_kernel(const float* __restrict__ xin, const float* __restrict__ delta,
              const float* __restrict__ g, const float* __restrict__ bv,
              float* __restrict__ xout, _Float16* __restrict__ x16)
{
    int row = blockIdx.x, t = threadIdx.x;
    __shared__ float red[256];
    long long o = (long long)row * DD + t;
    float v = xin[o] + delta[o];
    red[t] = v; __syncthreads();
    for (int s = 128; s > 0; s >>= 1) { if (t < s) red[t] += red[t + s]; __syncthreads(); }
    float mu = red[0] * (1.f / 256.f);
    __syncthreads();
    float dv = v - mu;
    red[t] = dv * dv; __syncthreads();
    for (int s = 128; s > 0; s >>= 1) { if (t < s) red[t] += red[t + s]; __syncthreads(); }
    float y = dv * rsqrtf(red[0] * (1.f / 256.f) + 1e-5f) * g[t] + bv[t];
    xout[o] = y;
    x16[o] = (_Float16)y;
}

__global__ void finalout_kernel(const float* __restrict__ tok, float* __restrict__ out,
                                long long total)
{
    long long i = (long long)blockIdx.x * blockDim.x + threadIdx.x;
    if (i >= total) return;                       // total = NB*OUTTOK*DD
    int d = (int)(i & 255);
    long long t = i >> 8;
    int rr = (int)(t % OUTTOK), b = (int)(t / OUTTOK);
    int src = (rr < NOBJ * TT) ? rr : (NOBJ * TT + (rr - NOBJ * TT) / TT);
    out[i] = tok[((long long)b * NTOK + src) * DD + d];
}

// ---------------------------------------------------------------------------
// Host launcher
// ---------------------------------------------------------------------------
extern "C" void kernel_launch(void* const* d_in, const int* in_sizes, int n_in,
                              void* d_out, int out_size, void* d_ws, size_t ws_size,
                              hipStream_t stream)
{
    // params flattened in sorted key order (jax pytree), then tensors in dict order
    const float* ag_mlp_b0 = (const float*)d_in[0];
    const float* ag_mlp_b1 = (const float*)d_in[1];
    const float* ag_mlp_w0 = (const float*)d_in[2];
    const float* ag_mlp_w1 = (const float*)d_in[3];
    const float* ag_out_b1 = (const float*)d_in[4];
    const float* ag_out_b2 = (const float*)d_in[5];
    const float* ag_out_w1 = (const float*)d_in[6];
    const float* ag_out_w2 = (const float*)d_in[7];
    const float* ag_pre_b0 = (const float*)d_in[8];
    const float* ag_pre_w0 = (const float*)d_in[9];
    const float* bk        = (const float*)d_in[10];
    const float* bo        = (const float*)d_in[11];
    const float* bq        = (const float*)d_in[12];
    const float* bv        = (const float*)d_in[13];
    const float* ffn_b1    = (const float*)d_in[14];
    const float* ffn_b2    = (const float*)d_in[15];
    const float* ffn_w1    = (const float*)d_in[16];
    const float* ffn_w2    = (const float*)d_in[17];
    const float* ln1_b     = (const float*)d_in[18];
    const float* ln1_g     = (const float*)d_in[19];
    const float* ln2_b     = (const float*)d_in[20];
    const float* ln2_g     = (const float*)d_in[21];
    const float* mp_mlp_b0 = (const float*)d_in[22];
    const float* mp_mlp_b1 = (const float*)d_in[23];
    const float* mp_mlp_w0 = (const float*)d_in[24];
    const float* mp_mlp_w1 = (const float*)d_in[25];
    const float* mp_out_b1 = (const float*)d_in[26];
    const float* mp_out_b2 = (const float*)d_in[27];
    const float* mp_out_w1 = (const float*)d_in[28];
    const float* mp_out_w2 = (const float*)d_in[29];
    const float* mp_pre_b0 = (const float*)d_in[30];
    const float* mp_pre_b1 = (const float*)d_in[31];
    const float* mp_pre_b2 = (const float*)d_in[32];
    const float* mp_pre_w0 = (const float*)d_in[33];
    const float* mp_pre_w1 = (const float*)d_in[34];
    const float* mp_pre_w2 = (const float*)d_in[35];
    const float* wk        = (const float*)d_in[36];
    const float* wo        = (const float*)d_in[37];
    const float* wq        = (const float*)d_in[38];
    const float* wv        = (const float*)d_in[39];
    const float* obj_trajs   = (const float*)d_in[40];
    const float* obj_pos     = (const float*)d_in[41];
    const float* map_polys   = (const float*)d_in[42];
    const float* map_center  = (const float*)d_in[43];
    const unsigned char* obj_mask = (const unsigned char*)d_in[44];
    const unsigned char* map_mask = (const unsigned char*)d_in[45];

    char* base = (char*)d_ws;
    size_t off = 0;
    auto bump = [&](size_t bytes) -> void* {
        off = (off + 255) & ~(size_t)255;
        void* p = base + off;
        off += bytes;
        return p;
    };
    // weights (f16, transposed [N][Kpad])
    _Float16* agPreT  = (_Float16*)bump((size_t)256*32*2);
    _Float16* agM0T   = (_Float16*)bump((size_t)256*512*2);
    _Float16* agM1T   = (_Float16*)bump((size_t)256*256*2);
    _Float16* agO1T   = (_Float16*)bump((size_t)256*256*2);
    _Float16* agO2T   = (_Float16*)bump((size_t)256*256*2);
    _Float16* mpP0T   = (_Float16*)bump((size_t)NTY*64*32*2);
    _Float16* mpP1T   = (_Float16*)bump((size_t)NTY*64*64*2);
    _Float16* mpP2T   = (_Float16*)bump((size_t)NTY*64*64*2);
    _Float16* mpM0T   = (_Float16*)bump((size_t)NTY*64*128*2);
    _Float16* mpM1T   = (_Float16*)bump((size_t)NTY*64*64*2);
    _Float16* mpO1T   = (_Float16*)bump((size_t)NTY*64*64*2);
    _Float16* mpO2T   = (_Float16*)bump((size_t)NTY*256*64*2);
    _Float16* wqT     = (_Float16*)bump((size_t)LL*256*256*2);
    _Float16* wkT     = (_Float16*)bump((size_t)LL*256*256*2);
    _Float16* wvT     = (_Float16*)bump((size_t)LL*256*256*2);
    _Float16* woT     = (_Float16*)bump((size_t)LL*256*256*2);
    _Float16* f1T     = (_Float16*)bump((size_t)LL*1024*256*2);
    _Float16* f2T     = (_Float16*)bump((size_t)LL*256*1024*2);
    // agent activations
    _Float16* Xa   = (_Float16*)bump((size_t)AGROWS*32*2);
    float*    am   = (float*)   bump((size_t)AGROWS*4);
    _Float16* H0   = (_Float16*)bump((size_t)AGROWS*256*2);
    _Float16* Hp   = (_Float16*)bump((size_t)NB*NOBJ*256*2);
    _Float16* Xc   = (_Float16*)bump((size_t)AGROWS*512*2);
    _Float16* H1   = (_Float16*)bump((size_t)AGROWS*256*2);
    _Float16* H2   = (_Float16*)bump((size_t)AGROWS*256*2);
    _Float16* H3   = (_Float16*)bump((size_t)AGROWS*256*2);
    float*    objf = (float*)   bump((size_t)AGROWS*256*4);
    // map activations (S0/S1 reused across stages)
    _Float16* Xm   = (_Float16*)bump((size_t)NTY*MPROWS*32*2);
    float*    mm   = (float*)   bump((size_t)NTY*MPROWS*4);
    float*    mv   = (float*)   bump((size_t)NTY*NB*PERT*4);
    _Float16* S0   = (_Float16*)bump((size_t)NTY*MPROWS*64*2);   // M0 -> M2 -> M4
    _Float16* S1   = (_Float16*)bump((size_t)NTY*MPROWS*64*2);   // M1 -> M3
    _Float16* Mp   = (_Float16*)bump((size_t)NTY*NB*PERT*64*2);
    _Float16* C    = (_Float16*)bump((size_t)NTY*MPROWS*128*2);  // Mcat; later ffn hidden
    _Float16* Mft  = (_Float16*)bump((size_t)NTY*NB*PERT*64*2);
    _Float16* M5   = (_Float16*)bump((size_t)NTY*NB*PERT*64*2);
    float*    mapf = (float*)   bump((size_t)NTY*NB*PERT*256*4);
    // transformer
    float*    tok  = (float*)   bump((size_t)TROWS*DD*4);
    float*    tm   = (float*)   bump((size_t)TROWS*4);
    float*    pos  = (float*)   bump((size_t)TROWS*DD*4);
    _Float16* x16  = (_Float16*)bump((size_t)TROWS*DD*2);
    _Float16* q16  = (_Float16*)bump((size_t)TROWS*DD*2);
    _Float16* Q16  = (_Float16*)bump((size_t)TROWS*DD*2);
    _Float16* K16  = (_Float16*)bump((size_t)TROWS*DD*2);
    _Float16* V16  = (_Float16*)bump((size_t)TROWS*DD*2);
    _Float16* O16  = (_Float16*)bump((size_t)TROWS*DD*2);
    float*    proj = (float*)   bump((size_t)TROWS*DD*4);
    (void)ws_size; (void)in_sizes; (void)n_in; (void)out_size;

    auto blocks = [](long long total) { return (unsigned)((total + 255) / 256); };
    auto cvtW = [&](const float* W, _Float16* Wt, int K, int N, int Kp, int batch) {
        long long total = (long long)batch * N * Kp;
        convert_wT_kernel<<<blocks(total), 256, 0, stream>>>(W, Wt, K, N, Kp, total);
    };
    auto gemm = [&](bool relu, const _Float16* X, long long xs,
                    const _Float16* W, long long wsd,
                    const float* bias, long long bs,
                    const float* rm, long long ms,
                    float* oF, _Float16* oH, long long os,
                    int M, int N, int K, int batch) {
        dim3 g(M / 64, N / 64, batch);
        if (relu) gemm64_kernel<true ><<<g, 128, 0, stream>>>(X, xs, W, wsd, bias, bs, rm, ms, oF, oH, os, M, N, K);
        else      gemm64_kernel<false><<<g, 128, 0, stream>>>(X, xs, W, wsd, bias, bs, rm, ms, oF, oH, os, M, N, K);
    };

    // ---- weight conversion ----
    cvtW(ag_pre_w0, agPreT, 30, 256, 32, 1);
    cvtW(ag_mlp_w0, agM0T, 512, 256, 512, 1);
    cvtW(ag_mlp_w1, agM1T, 256, 256, 256, 1);
    cvtW(ag_out_w1, agO1T, 256, 256, 256, 1);
    cvtW(ag_out_w2, agO2T, 256, 256, 256, 1);
    cvtW(mp_pre_w0, mpP0T, 9, 64, 32, NTY);
    cvtW(mp_pre_w1, mpP1T, 64, 64, 64, NTY);
    cvtW(mp_pre_w2, mpP2T, 64, 64, 64, NTY);
    cvtW(mp_mlp_w0, mpM0T, 128, 64, 128, NTY);
    cvtW(mp_mlp_w1, mpM1T, 64, 64, 64, NTY);
    cvtW(mp_out_w1, mpO1T, 64, 64, 64, NTY);
    cvtW(mp_out_w2, mpO2T, 64, 256, 64, NTY);
    cvtW(wq, wqT, 256, 256, 256, LL);
    cvtW(wk, wkT, 256, 256, 256, LL);
    cvtW(wv, wvT, 256, 256, 256, LL);
    cvtW(wo, woT, 256, 256, 256, LL);
    cvtW(ffn_w1, f1T, 256, 1024, 256, LL);
    cvtW(ffn_w2, f2T, 1024, 256, 1024, LL);

    // ---- agent pointnet ----
    pack_agent_kernel<<<blocks((long long)AGROWS*32), 256, 0, stream>>>(obj_trajs, obj_mask, Xa, am, (long long)AGROWS*32);
    gemm(true,  Xa, 0, agPreT, 0, ag_pre_b0, 0, am, 0, nullptr, H0, 0, AGROWS, 256, 32, 1);
    poolmax_kernel<<<blocks((long long)NB*NOBJ*256), 256, 0, stream>>>(H0, Hp, TT, 256, (long long)NB*NOBJ*256);
    concat_kernel<<<blocks((long long)AGROWS*512), 256, 0, stream>>>(H0, Hp, Xc, TT, 256, (long long)AGROWS*512);
    gemm(true,  Xc, 0, agM0T, 0, ag_mlp_b0, 0, nullptr, 0, nullptr, H1, 0, AGROWS, 256, 512, 1);
    gemm(true,  H1, 0, agM1T, 0, ag_mlp_b1, 0, am, 0, nullptr, H2, 0, AGROWS, 256, 256, 1);
    gemm(true,  H2, 0, agO1T, 0, ag_out_b1, 0, nullptr, 0, nullptr, H3, 0, AGROWS, 256, 256, 1);
    gemm(false, H3, 0, agO2T, 0, ag_out_b2, 0, am, 0, objf, nullptr, 0, AGROWS, 256, 256, 1);
    scatter_obj_kernel<<<blocks((long long)AGROWS*256), 256, 0, stream>>>(objf, tok, (long long)AGROWS*256);

    // ---- map pointnet (batched over 16 types) ----
    pack_map_kernel<<<blocks((long long)NTY*MPROWS*32), 256, 0, stream>>>(map_polys, map_mask, Xm, mm, (long long)NTY*MPROWS*32);
    mvalid_kernel<<<blocks((long long)NTY*NB*PERT), 256, 0, stream>>>(mm, mv, (long long)NTY*NB*PERT);
    gemm(true, Xm, (long long)MPROWS*32,  mpP0T, 64*32,  mp_pre_b0, 64, nullptr, 0, nullptr, S0, (long long)MPROWS*64, MPROWS, 64, 32, NTY);
    gemm(true, S0, (long long)MPROWS*64,  mpP1T, 64*64,  mp_pre_b1, 64, nullptr, 0, nullptr, S1, (long long)MPROWS*64, MPROWS, 64, 64, NTY);
    gemm(true, S1, (long long)MPROWS*64,  mpP2T, 64*64,  mp_pre_b2, 64, mm, MPROWS, nullptr, S0, (long long)MPROWS*64, MPROWS, 64, 64, NTY);
    poolmax_kernel<<<blocks((long long)NTY*NB*PERT*64), 256, 0, stream>>>(S0, Mp, NPTS, 64, (long long)NTY*NB*PERT*64);
    concat_kernel<<<blocks((long long)NTY*MPROWS*128), 256, 0, stream>>>(S0, Mp, C, NPTS, 64, (long long)NTY*MPROWS*128);
    gemm(true, C,  (long long)MPROWS*128, mpM0T, 64*128, mp_mlp_b0, 64, nullptr, 0, nullptr, S1, (long long)MPROWS*64, MPROWS, 64, 128, NTY);
    gemm(true, S1, (long long)MPROWS*64,  mpM1T, 64*64,  mp_mlp_b1, 64, mm, MPROWS, nullptr, S0, (long long)MPROWS*64, MPROWS, 64, 64, NTY);
    poolmax_kernel<<<blocks((long long)NTY*NB*PERT*64), 256, 0, stream>>>(S0, Mft, NPTS, 64, (long long)NTY*NB*PERT*64);
    gemm(true,  Mft, (long long)NB*PERT*64, mpO1T, 64*64,  mp_out_b1, 64,  nullptr, 0, nullptr, M5, (long long)NB*PERT*64, NB*PERT, 64, 64, NTY);
    gemm(false, M5,  (long long)NB*PERT*64, mpO2T, 256*64, mp_out_b2, 256, mv, NB*PERT, mapf, nullptr, (long long)NB*PERT*256, NB*PERT, 256, 64, NTY);
    scatter_map_kernel<<<blocks((long long)NTY*NB*PERT*256), 256, 0, stream>>>(mapf, tok, (long long)NTY*NB*PERT*256);

    // ---- token mask & positional embedding ----
    tokmask_kernel<<<blocks((long long)TROWS), 256, 0, stream>>>(am, mv, tm, (long long)TROWS);
    posembed_kernel<<<blocks((long long)TROWS*DD), 256, 0, stream>>>(obj_pos, map_center, pos, (long long)TROWS*DD);

    // ---- transformer layers ----
    _Float16* ffnh = C;   // reuse big map-concat region for FFN hidden
    for (int i = 0; i < LL; ++i) {
        prep_kernel<<<blocks((long long)TROWS*DD), 256, 0, stream>>>(tok, pos, x16, q16, (long long)TROWS*DD);
        gemm(false, q16, 0, wqT + (size_t)i*256*256, 0, bq + i*256, 0, nullptr, 0, nullptr, Q16, 0, TROWS, 256, 256, 1);
        gemm(false, q16, 0, wkT + (size_t)i*256*256, 0, bk + i*256, 0, nullptr, 0, nullptr, K16, 0, TROWS, 256, 256, 1);
        gemm(false, x16, 0, wvT + (size_t)i*256*256, 0, bv + i*256, 0, nullptr, 0, nullptr, V16, 0, TROWS, 256, 256, 1);
        attn_kernel<<<dim3(NTOK/16, HH, NB), 32, 0, stream>>>(Q16, K16, V16, tm, O16);
        gemm(false, O16, 0, woT + (size_t)i*256*256, 0, bo + i*256, 0, nullptr, 0, proj, nullptr, 0, TROWS, 256, 256, 1);
        add_ln_kernel<<<TROWS, 256, 0, stream>>>(tok, proj, ln1_g + i*256, ln1_b + i*256, tok, x16);
        gemm(true,  x16,  0, f1T + (size_t)i*1024*256, 0, ffn_b1 + i*1024, 0, nullptr, 0, nullptr, ffnh, 0, TROWS, 1024, 256, 1);
        gemm(false, ffnh, 0, f2T + (size_t)i*256*1024, 0, ffn_b2 + i*256,  0, nullptr, 0, proj, nullptr, 0, TROWS, 256, 1024, 1);
        add_ln_kernel<<<TROWS, 256, 0, stream>>>(tok, proj, ln2_g + i*256, ln2_b + i*256, tok, x16);
    }

    // ---- output assembly (obj tokens + map tokens broadcast over T) ----
    finalout_kernel<<<blocks((long long)NB*OUTTOK*DD), 256, 0, stream>>>(tok, (float*)d_out, (long long)NB*OUTTOK*DD);
}
